// VectorQuantizer_53300544143963
// MI455X (gfx1250) — matrix operand order repaired and checked by reading, not compiled, not run on hardware
//
#include <hip/hip_runtime.h>
#include <hip/hip_bf16.h>

// ---------------------------------------------------------------------------
// VQ-VAE vector quantizer for MI455X (gfx1250, wave32, WMMA)
//   inputs  : [16, 16384, 64] f32   (N = 262144 rows, D = 64)
//   codebook: [512, 64] f32         (K = 512 codes)
//   out     : [1 + N*64] f32  = { loss, quantized }
// scores = codebook x inputs^T via v_wmma_f32_16x16x32_f16, with the
// -0.5*||e||^2 bias pre-negated in LDS and fed as the WMMA C operand.
// Each wave processes 4 row tiles concurrently so the A-fragments and the
// C-init (both shared across row tiles) are loaded from LDS once per
// 8 WMMAs -> LDS traffic 4x below the naive version, and 4 independent
// accumulator chains keep the matrix pipe busy. Output gather is exact
// f32 codebook values, fully coalesced.
// ---------------------------------------------------------------------------

typedef _Float16 half_t;
typedef __attribute__((ext_vector_type(16))) _Float16 v16h;
typedef __attribute__((ext_vector_type(8)))  float    v8f;

#define NUM_CODES   512
#define DIM         64
#define N_ROWS      262144      // 16 * 16384
#define CB_PITCH    72          // halves per LDS codebook row (144B: conflict-free b128)
#define WAVES       8           // waves per block (256 threads)
#define MT          4           // row tiles processed concurrently per wave
#define OUTER       2           // tile groups per wave (total 8 row tiles/wave)
#define GRID_BLOCKS (N_ROWS / (16 * WAVES * MT * OUTER))       // 256
#define SMEM_BYTES  (NUM_CODES * CB_PITCH * 2 + NUM_CODES * 4) // 75776

__global__ void vq_init_kernel(float* acc) { acc[0] = 0.0f; }

__global__ void vq_finalize_kernel(const float* acc, float* out) {
    // loss = q_latent + 0.25 * e_latent = 1.25 * mean(diff^2)
    out[0] = acc[0] * (1.25f / 16777216.0f);
}

__global__ void __launch_bounds__(256)
vq_main_kernel(const float* __restrict__ x,      // [N_ROWS, 64]
               const float* __restrict__ cb,     // [512, 64]
               float* __restrict__ outq,         // [N_ROWS, 64] (d_out + 1)
               float* __restrict__ lossAcc)      // scratch accumulator
{
    extern __shared__ char smem_raw[];
    half_t* cbH = (half_t*)smem_raw;                               // 512 x CB_PITCH f16
    float*  hn  = (float*)(smem_raw + NUM_CODES * CB_PITCH * 2);   // -0.5*||e||^2

    const int tid = threadIdx.x;

    // ---- stage codebook to LDS as f16 (padded pitch) + negated half-norms ----
    for (int i = tid; i < NUM_CODES * DIM; i += 256) {
        int r = i >> 6, c = i & 63;
        cbH[r * CB_PITCH + c] = (half_t)cb[i];
    }
    for (int r = tid; r < NUM_CODES; r += 256) {
        float s = 0.0f;
        const float* row = cb + r * DIM;
#pragma unroll
        for (int c = 0; c < DIM; ++c) { float v = row[c]; s += v * v; }
        hn[r] = -0.5f * s;       // pre-negated: feeds WMMA C operand directly
    }
    __syncthreads();

    const int wave = tid >> 5;
    const int lane = tid & 31;
    const int llow = lane & 15;   // row / code-row within half
    const int lhi  = lane >> 4;   // 0 or 1 (K-half / M-half selector)

    float ssq = 0.0f;             // per-lane squared-error partial

    for (int g = 0; g < OUTER; ++g) {
        const int rowBase0 =
            ((blockIdx.x * WAVES + wave) * OUTER + g) * (MT * 16);

        // ---- B fragments for MT row tiles: inputs^T (K = dim, N = row) ----
        // lane holds column N = llow, K = kb*32 + lhi*16 + h (16 contiguous f32)
        v16h b0[MT], b1[MT];
#pragma unroll
        for (int m = 0; m < MT; ++m) {
            const float* xr = x + (size_t)(rowBase0 + m * 16 + llow) * DIM + lhi * 16;
#pragma unroll
            for (int h = 0; h < 16; ++h) b0[m][h] = (half_t)xr[h];
#pragma unroll
            for (int h = 0; h < 16; ++h) b1[m][h] = (half_t)xr[32 + h];
        }

        float bestScore[MT];
        int   bestIdx[MT];
#pragma unroll
        for (int m = 0; m < MT; ++m) { bestScore[m] = -3.402823466e+38f; bestIdx[m] = 0; }

        // ---- sweep 32 code tiles of 16 codes each ----
        for (int ct = 0; ct < 32; ++ct) {
            const int cbase = ct * 16;

            // shared C init: cinit[r] = -0.5*||e||^2, code m = cbase + lhi*8 + r
            // (two 32B-aligned ds_load_b128, reused by all MT accumulators)
            const float4* hp4 = (const float4*)(hn + cbase + lhi * 8);
            float4 h0 = hp4[0], h1 = hp4[1];
            v8f cinit;
            cinit[0] = h0.x; cinit[1] = h0.y; cinit[2] = h0.z; cinit[3] = h0.w;
            cinit[4] = h1.x; cinit[5] = h1.y; cinit[6] = h1.z; cinit[7] = h1.w;

            // shared A fragments: code row m = cbase + llow,
            // K(h) = lhi*16 + h  (+32 for second fragment)
            const half_t* ar = cbH + (size_t)(cbase + llow) * CB_PITCH + lhi * 16;
            v16h a0, a1;
#pragma unroll
            for (int h = 0; h < 8; ++h) { a0[h] = ar[h];      a0[8 + h] = ar[8 + h]; }
#pragma unroll
            for (int h = 0; h < 8; ++h) { a1[h] = ar[32 + h]; a1[8 + h] = ar[40 + h]; }

            v8f acc[MT];
#pragma unroll
            for (int m = 0; m < MT; ++m) {
                acc[m] = __builtin_amdgcn_wmma_f32_16x16x32_f16(
                    false, a0, false, b0[m], (short)0, cinit, false, false);
            }
#pragma unroll
            for (int m = 0; m < MT; ++m) {
                acc[m] = __builtin_amdgcn_wmma_f32_16x16x32_f16(
                    false, a1, false, b1[m], (short)0, acc[m], false, false);
            }

            // acc[m][r] = x_row . e_c - 0.5*||e_c||^2, c = cbase + lhi*8 + r
#pragma unroll
            for (int m = 0; m < MT; ++m) {
#pragma unroll
                for (int r = 0; r < 8; ++r) {
                    float s = acc[m][r];
                    int   c = cbase + lhi * 8 + r;
                    if (s > bestScore[m]) { bestScore[m] = s; bestIdx[m] = c; }
                }
            }
        }

        // ---- per row tile: combine lane-halves, then coalesced output ----
#pragma unroll
        for (int m = 0; m < MT; ++m) {
            float bS = bestScore[m];
            int   bI = bestIdx[m];
            float oS = __shfl_xor(bS, 16, 32);
            int   oI = __shfl_xor(bI, 16, 32);
            if (oS > bS || (oS == bS && oI < bI)) { bS = oS; bI = oI; }
            // lanes 0..15 (mirrored in 16..31) hold argmin for row rowBase+llow

            const int    rowBase = rowBase0 + m * 16;
            const size_t outBase = (size_t)rowBase * DIM;
#pragma unroll 4
            for (int k = 0; k < 32; ++k) {
                int row = k >> 1;                     // uniform across the wave
                int idx = __shfl(bI, row, 32);
                int col = ((k & 1) << 5) + lane;
                float q  = cb[idx * DIM + col];       // exact f32 codebook gather
                size_t o = outBase + (size_t)row * DIM + col;
                float xv = x[o];
                outq[o]  = q;
                float d  = q - xv;
                ssq += d * d;
            }
        }
    }

    // ---- wave reduction + one atomic per wave ----
#pragma unroll
    for (int off = 16; off > 0; off >>= 1) ssq += __shfl_xor(ssq, off, 32);
    if (lane == 0) atomicAdd(lossAcc, ssq);
}

extern "C" void kernel_launch(void* const* d_in, const int* in_sizes, int n_in,
                              void* d_out, int out_size, void* d_ws, size_t ws_size,
                              hipStream_t stream) {
    const float* x  = (const float*)d_in[0];   // inputs  [16,16384,64]
    const float* cb = (const float*)d_in[1];   // codebook [512,64]
    float* out      = (float*)d_out;           // [0]=loss, [1..]=quantized
    float* lossAcc  = (float*)d_ws;

    (void)in_sizes; (void)n_in; (void)out_size; (void)ws_size;

    // allow >64KB dynamic LDS (WGP supports up to 320KB)
    (void)hipFuncSetAttribute((const void*)vq_main_kernel,
                              hipFuncAttributeMaxDynamicSharedMemorySize,
                              SMEM_BYTES);

    vq_init_kernel<<<1, 1, 0, stream>>>(lossAcc);
    vq_main_kernel<<<GRID_BLOCKS, 256, SMEM_BYTES, stream>>>(x, cb, out + 1, lossAcc);
    vq_finalize_kernel<<<1, 1, 0, stream>>>(lossAcc, out);
}
